// LLaMALayer_64381559767430
// MI455X (gfx1250) — matrix operand order repaired and checked by reading, not compiled, not run on hardware
//
#include <hip/hip_runtime.h>
#include <hip/hip_bf16.h>

typedef __attribute__((ext_vector_type(16))) __bf16        v16bf;
typedef __attribute__((ext_vector_type(8)))  float         v8f;
typedef __attribute__((ext_vector_type(4)))  unsigned int  u32x4;
typedef __attribute__((ext_vector_type(4)))  float         f32x4;
typedef unsigned short u16;
typedef unsigned int   u32;

#define HIDDEN 4096
#define FFN    11008
#define NHEADS 32
#define HDIM   128
#define SEQ    1024
#define PASTN  1024

// -------- CDNA5 async global->LDS path (guarded; falls back to sync copy) ----
#if defined(__gfx1250__) && __has_builtin(__builtin_amdgcn_global_load_async_to_lds_b128)
#define USE_ASYNC_LDS 1
#else
#define USE_ASYNC_LDS 0
#endif

#if USE_ASYNC_LDS
// builtin expects: (v4i addrspace(1)* src, v4i addrspace(3)* dst, imm offset, imm cpol)
typedef int v4i __attribute__((vector_size(16)));
typedef __attribute__((address_space(1))) v4i as1_v4i;
typedef __attribute__((address_space(3))) v4i as3_v4i;

__device__ __forceinline__ void async_cp16(u16* lds_dst, const u16* gsrc) {
    // AS3 pointer rebuilt from the low 32 bits of the flat address (= LDS offset)
    __builtin_amdgcn_global_load_async_to_lds_b128(
        (as1_v4i*)(uintptr_t)gsrc,
        (as3_v4i*)(u32)(uintptr_t)lds_dst,
        0, 0);
}
__device__ __forceinline__ void wait_async0() {
#if __has_builtin(__builtin_amdgcn_s_wait_asynccnt)
    __builtin_amdgcn_s_wait_asynccnt(0);
#else
    asm volatile("s_wait_asynccnt 0x0" ::: "memory");
#endif
}
#endif

// ---------- helpers ----------
__device__ __forceinline__ u16 f2bf(float f) {
    u32 u = __float_as_uint(f);
    u = (u + 0x7FFFu + ((u >> 16) & 1u)) >> 16;   // round-to-nearest-even
    return (u16)u;
}

union BFrag { u32x4 q[2]; v16bf v; };

// build a 16-element bf16 WMMA fragment from two 16-byte chunks
__device__ __forceinline__ v16bf ld_frag_pair(const u16* p0, const u16* p1) {
    BFrag f;
    f.q[0] = *(const u32x4*)p0;
    f.q[1] = *(const u32x4*)p1;
    return f.v;
}

__device__ __forceinline__ v8f vzero8() {
    v8f v;
#pragma unroll
    for (int i = 0; i < 8; ++i) v[i] = 0.0f;
    return v;
}

// ---------- fp32 -> bf16 weight conversion (one streaming pass) ----------
__global__ void cvt_f32_to_bf16(const float* __restrict__ in, u16* __restrict__ out, int n) {
    int i = (blockIdx.x * blockDim.x + threadIdx.x) * 4;
    if (i >= n) return;
    f32x4 v = *(const f32x4*)(in + i);
    u32 lo = (u32)f2bf(v[0]) | ((u32)f2bf(v[1]) << 16);
    u32 hi = (u32)f2bf(v[2]) | ((u32)f2bf(v[3]) << 16);
    *(u32*)(out + i)     = lo;
    *(u32*)(out + i + 2) = hi;
}

// ---------- RMSNorm (fp32 in, bf16 out) ----------
__global__ void __launch_bounds__(256)
rmsnorm_to_bf16(const float* __restrict__ X, const float* __restrict__ W, u16* __restrict__ O) {
    __shared__ float red[8];
    const int row = blockIdx.x;
    const float* x = X + (size_t)row * HIDDEN;
    float ss = 0.0f;
    for (int i = threadIdx.x; i < HIDDEN; i += 256) { float v = x[i]; ss += v * v; }
#pragma unroll
    for (int off = 16; off; off >>= 1) ss += __shfl_xor(ss, off, 32);
    if ((threadIdx.x & 31) == 0) red[threadIdx.x >> 5] = ss;
    __syncthreads();
    float tot = 0.0f;
#pragma unroll
    for (int i = 0; i < 8; ++i) tot += red[i];
    float sc = rsqrtf(tot * (1.0f / HIDDEN) + 1e-6f);
    u16* out = O + (size_t)row * HIDDEN;
    for (int i = threadIdx.x; i < HIDDEN; i += 256) out[i] = f2bf(x[i] * sc * W[i]);
}

// ---------- RoPE (fp32 in, bf16 out), matches xe/xo-interleave -> concat ----------
__global__ void rope_to_bf16(const float* __restrict__ X, u16* __restrict__ O) {
    int t = blockIdx.x * blockDim.x + threadIdx.x;
    if (t >= SEQ * NHEADS * (HDIM / 2)) return;
    int i = t & 63;
    int h = (t >> 6) & 31;
    int s = t >> 11;
    const float* xp = X + (size_t)s * HIDDEN + h * HDIM;
    float xe = xp[2 * i], xo = xp[2 * i + 1];
    float fr = exp2f((-2.0f * (float)i / (float)HDIM) * 13.28771237954945f); // log2(10000)
    float ang = (float)s * fr;
    float sn, cs;
    sincosf(ang, &sn, &cs);
    u16* op = O + (size_t)s * HIDDEN + h * HDIM;
    op[i]      = f2bf(xe * cs - xo * sn);
    op[64 + i] = f2bf(xe * sn + xo * cs);
}

// ---------- SwiGLU: g = x1 * silu(x3), bf16 out ----------
__global__ void swiglu_to_bf16(const float* __restrict__ X1, const float* __restrict__ X3,
                               u16* __restrict__ G, int n) {
    int i = blockIdx.x * blockDim.x + threadIdx.x;
    if (i >= n) return;
    float a = X1[i], b = X3[i];
    float s = b / (1.0f + __expf(-b));
    G[i] = f2bf(a * s);
}

// ---------- Generic bf16 WMMA GEMM: C[M,N] = A[M,K] * B[N,K]^T ----------
// Block: 256 threads (8 waves), 128x128 tile, K-step 32, double-buffered LDS
// with async global->LDS staging. Wave tile: 64(M) x 32(N) => 8 WMMAs/K-step.
#define LDT 40   // LDS row stride in halves (80B: 16B aligned, bank-spread)
__global__ void __launch_bounds__(256)
gemm_bf16_wmma(const u16* __restrict__ A, const u16* __restrict__ B,
               float* __restrict__ C, int M, int N, int K) {
    __shared__ u16 As[2][128 * LDT];
    __shared__ u16 Bs[2][128 * LDT];
    const int tid  = threadIdx.x;
    const int lane = tid & 31;
    const int wid  = tid >> 5;
    const int lh = lane & 15;
    const int hi = lane >> 4;
    const int wm = (wid & 1) * 64;   // wave M offset in block tile
    const int wn = (wid >> 1) * 32;  // wave N offset in block tile
    const int row0 = blockIdx.y * 128;
    const int col0 = blockIdx.x * 128;

    v8f c[4][2];
#pragma unroll
    for (int f = 0; f < 4; ++f)
#pragma unroll
        for (int g = 0; g < 2; ++g) c[f][g] = vzero8();

    // stage one 128x32 bf16 tile of A and B into LDS buffer `buf`
    auto stage = [&](int kt, int buf) {
#pragma unroll
        for (int i = 0; i < 2; ++i) {
            int idx = tid + i * 256;          // 0..511
            int rr  = idx >> 2;               // 0..127
            int kc  = (idx & 3) * 8;          // 0,8,16,24 (halves)
            const u16* ga = &A[(size_t)(row0 + rr) * K + kt + kc];
            const u16* gb = &B[(size_t)(col0 + rr) * K + kt + kc];
            u16* la = &As[buf][rr * LDT + kc];
            u16* lb = &Bs[buf][rr * LDT + kc];
#if USE_ASYNC_LDS
            async_cp16(la, ga);
            async_cp16(lb, gb);
#else
            *(u32x4*)la = *(const u32x4*)ga;
            *(u32x4*)lb = *(const u32x4*)gb;
#endif
        }
    };

    const int nk = K >> 5;
    stage(0, 0);
#if USE_ASYNC_LDS
    wait_async0();
#endif
    __syncthreads();

    for (int t = 0; t < nk; ++t) {
        const int cur = t & 1;
        if (t + 1 < nk) stage((t + 1) << 5, cur ^ 1);   // overlap next tile

        // A frag: lane lh holds row, K-chunks [hi*8 .. +7] and [hi*8+16 .. +7]
        v16bf a[4], b[2];
#pragma unroll
        for (int f = 0; f < 4; ++f) {
            const u16* p = &As[cur][(wm + f * 16 + lh) * LDT + hi * 8];
            a[f] = ld_frag_pair(p, p + 16);
        }
        // B frag: lane lh holds col n, K-chunks [hi*8 .. +7] and [hi*8+16 .. +7]
#pragma unroll
        for (int g = 0; g < 2; ++g) {
            const u16* p = &Bs[cur][(wn + g * 16 + lh) * LDT + hi * 8];
            b[g] = ld_frag_pair(p, p + 16);
        }
#pragma unroll
        for (int f = 0; f < 4; ++f)
#pragma unroll
            for (int g = 0; g < 2; ++g)
                c[f][g] = __builtin_amdgcn_wmma_f32_16x16x32_bf16(
                    false, a[f], false, b[g], (short)0, c[f][g], false, false);

#if USE_ASYNC_LDS
        wait_async0();            // our async writes into the other buffer done
#endif
        __syncthreads();          // everyone finished reading `cur` + tile ready
    }

    // C frag layout: VGPR r -> M = r + 8*hi ; N = lane&15
#pragma unroll
    for (int f = 0; f < 4; ++f)
#pragma unroll
        for (int g = 0; g < 2; ++g)
#pragma unroll
            for (int r = 0; r < 8; ++r) {
                int row = row0 + wm + f * 16 + r + hi * 8;
                int col = col0 + wn + g * 16 + lh;
                C[(size_t)row * N + col] = c[f][g][r];
            }
}

// ---------- Flash attention (no scale, matching reference softmax(q·kᵀ)) ----------
// grid: (NHEADS, SEQ/64); block 128 = 4 waves; each wave owns 16 q rows.
#define KLD 136   // K-tile LDS row stride (halves)
__global__ void __launch_bounds__(128)
attn_flash_wmma(const u16* __restrict__ Q, const float* __restrict__ PK,
                const float* __restrict__ PV, u16* __restrict__ O) {
    __shared__ u16 Ks[32 * KLD];        // K tile [p][d], d contiguous
    __shared__ u16 Vt[128 * LDT];       // V tile transposed [d][p]
    __shared__ u16 Pst[4 * 16 * LDT];   // per-wave P transpose scratch
    const int tid  = threadIdx.x;
    const int lane = tid & 31;
    const int w    = tid >> 5;
    const int lh = lane & 15;
    const int hi = lane >> 4;
    const int head  = blockIdx.x;
    const int qrow0 = blockIdx.y * 64 + w * 16;

    // Q A-fragments: 16 rows x 128 dims = 4 K-chunks of 32
    v16bf qa[4];
#pragma unroll
    for (int kc = 0; kc < 4; ++kc) {
        const u16* p = Q + (size_t)(qrow0 + lh) * HIDDEN + head * HDIM + kc * 32 + hi * 8;
        qa[kc] = ld_frag_pair(p, p + 16);
    }

    v8f o[8];
#pragma unroll
    for (int d = 0; d < 8; ++d) o[d] = vzero8();
    float mrow[8], lrow[8];
#pragma unroll
    for (int r = 0; r < 8; ++r) { mrow[r] = -3.0e38f; lrow[r] = 0.0f; }

    u16* Pw = &Pst[w * 16 * LDT];
    const int sp = tid >> 2;        // staging: past row 0..31
    const int sd = (tid & 3) * 32;  // staging: dim base

    for (int pt = 0; pt < PASTN; pt += 32) {
        // stage K tile 32x128 (fp32 -> bf16)
        const float* ksrc = PK + ((size_t)head * PASTN + pt + sp) * HDIM + sd;
        const float* vsrc = PV + ((size_t)head * PASTN + pt + sp) * HDIM + sd;
#pragma unroll
        for (int j = 0; j < 8; ++j) {
            f32x4 v = *(const f32x4*)(ksrc + j * 4);
            *(u32*)&Ks[sp * KLD + sd + j * 4]     = (u32)f2bf(v[0]) | ((u32)f2bf(v[1]) << 16);
            *(u32*)&Ks[sp * KLD + sd + j * 4 + 2] = (u32)f2bf(v[2]) | ((u32)f2bf(v[3]) << 16);
        }
        // stage V transposed [d][p]
#pragma unroll
        for (int j = 0; j < 8; ++j) {
            f32x4 v = *(const f32x4*)(vsrc + j * 4);
#pragma unroll
            for (int e = 0; e < 4; ++e)
                Vt[(sd + j * 4 + e) * LDT + sp] = f2bf(v[e]);
        }
        __syncthreads();

        // raw scores for 32 past positions: two 16x16 fp32 frags
        v8f s0 = vzero8(), s1 = vzero8();
#pragma unroll
        for (int kc = 0; kc < 4; ++kc) {
            const u16* p0 = &Ks[(0 * 16 + lh) * KLD + kc * 32 + hi * 8];
            const u16* p1 = &Ks[(1 * 16 + lh) * KLD + kc * 32 + hi * 8];
            v16bf b0 = ld_frag_pair(p0, p0 + 16);
            v16bf b1 = ld_frag_pair(p1, p1 + 16);
            s0 = __builtin_amdgcn_wmma_f32_16x16x32_bf16(false, qa[kc], false, b0, (short)0, s0, false, false);
            s1 = __builtin_amdgcn_wmma_f32_16x16x32_bf16(false, qa[kc], false, b1, (short)0, s1, false, false);
        }

        // online softmax (row = r + 8*hi, cols across the 16-lane group)
#pragma unroll
        for (int r = 0; r < 8; ++r) {
            float a0 = s0[r], a1 = s1[r];
            float mx = fmaxf(a0, a1);
            mx = fmaxf(mx, __shfl_xor(mx, 1, 16));
            mx = fmaxf(mx, __shfl_xor(mx, 2, 16));
            mx = fmaxf(mx, __shfl_xor(mx, 4, 16));
            mx = fmaxf(mx, __shfl_xor(mx, 8, 16));
            float mn   = fmaxf(mrow[r], mx);
            float corr = __expf(mrow[r] - mn);
            float p0 = __expf(a0 - mn);
            float p1 = __expf(a1 - mn);
            float ps = p0 + p1;
            ps += __shfl_xor(ps, 1, 16);
            ps += __shfl_xor(ps, 2, 16);
            ps += __shfl_xor(ps, 4, 16);
            ps += __shfl_xor(ps, 8, 16);
            lrow[r] = lrow[r] * corr + ps;
            mrow[r] = mn;
#pragma unroll
            for (int d = 0; d < 8; ++d) o[d][r] = o[d][r] * corr;
            int m = r + hi * 8;
            Pw[m * LDT + lh]      = f2bf(p0);
            Pw[m * LDT + 16 + lh] = f2bf(p1);
        }
        asm volatile("s_wait_dscnt 0x0" ::: "memory");

        // P as A-fragment (16 x 32), then O += P x V
        const u16* pp = &Pw[lh * LDT + hi * 8];
        v16bf pa = ld_frag_pair(pp, pp + 16);
#pragma unroll
        for (int d = 0; d < 8; ++d) {
            const u16* vp = &Vt[(d * 16 + lh) * LDT + hi * 8];
            v16bf vb = ld_frag_pair(vp, vp + 16);
            o[d] = __builtin_amdgcn_wmma_f32_16x16x32_bf16(false, pa, false, vb, (short)0, o[d], false, false);
        }
        __syncthreads();
    }

    // normalize and write bf16 merged-head output
#pragma unroll
    for (int r = 0; r < 8; ++r) {
        float inv = 1.0f / lrow[r];
        int row = qrow0 + r + hi * 8;
#pragma unroll
        for (int d = 0; d < 8; ++d)
            O[(size_t)row * HIDDEN + head * HDIM + d * 16 + lh] = f2bf(o[d][r] * inv);
    }
}

// ---------- launch ----------
extern "C" void kernel_launch(void* const* d_in, const int* in_sizes, int n_in,
                              void* d_out, int out_size, void* d_ws, size_t ws_size,
                              hipStream_t stream) {
    (void)in_sizes; (void)n_in; (void)out_size; (void)ws_size;
    const float* x   = (const float*)d_in[0];
    const float* wq  = (const float*)d_in[1];
    // d_in[2]=wk, d_in[3]=wv are dead in the reference graph
    const float* wo  = (const float*)d_in[4];
    const float* w1  = (const float*)d_in[5];
    const float* w2  = (const float*)d_in[6];
    const float* w3  = (const float*)d_in[7];
    const float* anw = (const float*)d_in[8];
    const float* fnw = (const float*)d_in[9];
    const float* pk  = (const float*)d_in[10];
    const float* pv  = (const float*)d_in[11];
    float* out = (float*)d_out;

    char* ws = (char*)d_ws;
    size_t off = 0;
    auto alloc = [&](size_t b) { size_t r = off; off += (b + 255) & ~(size_t)255; return r; };

    const size_t HH  = (size_t)HIDDEN * HIDDEN;   // 16.7M
    const size_t FH  = (size_t)FFN * HIDDEN;      // 45.1M
    const size_t SH  = (size_t)SEQ * HIDDEN;      // 4.19M
    const size_t SF  = (size_t)SEQ * FFN;         // 11.3M

    u16* wq_bf = (u16*)(ws + alloc(HH * 2));
    u16* wo_bf = (u16*)(ws + alloc(HH * 2));
    u16* w1_bf = (u16*)(ws + alloc(FH * 2));
    u16* w3_bf = (u16*)(ws + alloc(FH * 2));
    u16* w2_bf = (u16*)(ws + alloc(FH * 2));
    u16* h_bf  = (u16*)(ws + alloc(SH * 2));      // reused for hf
    float* xq_f = (float*)(ws + alloc(SH * 4));   // reused for xo
    u16* q_bf  = (u16*)(ws + alloc(SH * 2));
    u16* at_bf = (u16*)(ws + alloc(SH * 2));
    float* x1_f = (float*)(ws + alloc(SF * 4));
    float* x3_f = (float*)(ws + alloc(SF * 4));
    u16* g_bf  = (u16*)(ws + alloc(SF * 2));

    auto cvt = [&](const float* src, u16* dst, size_t n) {
        int blocks = (int)((n / 4 + 255) / 256);
        cvt_f32_to_bf16<<<blocks, 256, 0, stream>>>(src, dst, (int)n);
    };
    cvt(wq, wq_bf, HH);
    cvt(wo, wo_bf, HH);
    cvt(w1, w1_bf, FH);
    cvt(w3, w3_bf, FH);
    cvt(w2, w2_bf, FH);

    // h = rmsnorm(x, attn_norm_w)
    rmsnorm_to_bf16<<<SEQ, 256, 0, stream>>>(x, anw, h_bf);

    // xq = h @ wq^T
    gemm_bf16_wmma<<<dim3(HIDDEN / 128, SEQ / 128), 256, 0, stream>>>(
        h_bf, wq_bf, xq_f, SEQ, HIDDEN, HIDDEN);

    // q = rope(xq) -> bf16
    rope_to_bf16<<<(SEQ * NHEADS * (HDIM / 2)) / 256, 256, 0, stream>>>(xq_f, q_bf);

    // attn over past_k/past_v (flash, online softmax)
    attn_flash_wmma<<<dim3(NHEADS, SEQ / 64), 128, 0, stream>>>(q_bf, pk, pv, at_bf);

    // xo = attn @ wo^T  (reuse xq_f buffer)
    gemm_bf16_wmma<<<dim3(HIDDEN / 128, SEQ / 128), 256, 0, stream>>>(
        at_bf, wo_bf, xq_f, SEQ, HIDDEN, HIDDEN);

    // hf = rmsnorm(xo, ffn_norm_w)  (reuse h_bf)
    rmsnorm_to_bf16<<<SEQ, 256, 0, stream>>>(xq_f, fnw, h_bf);

    // x1 = hf @ w1^T ; x3 = hf @ w3^T
    gemm_bf16_wmma<<<dim3(FFN / 128, SEQ / 128), 256, 0, stream>>>(
        h_bf, w1_bf, x1_f, SEQ, FFN, HIDDEN);
    gemm_bf16_wmma<<<dim3(FFN / 128, SEQ / 128), 256, 0, stream>>>(
        h_bf, w3_bf, x3_f, SEQ, FFN, HIDDEN);

    // g = x1 * silu(x3)
    swiglu_to_bf16<<<(int)((SF + 255) / 256), 256, 0, stream>>>(x1_f, x3_f, g_bf, (int)SF);

    // out = g @ w2^T  (K = 11008 = 344*32)
    gemm_bf16_wmma<<<dim3(HIDDEN / 128, SEQ / 128), 256, 0, stream>>>(
        g_bf, w2_bf, out, SEQ, HIDDEN, FFN);
}
